// biGRUCell_33784212750504
// MI455X (gfx1250) — compile-verified, matrix operand order checked by reading, not run on hardware
//
#include <hip/hip_runtime.h>
#include <cstdint>
#include <cstddef>

// Problem constants (from reference): B=128, T=512, I=512, H=1024, TGT=128
#define B_    128
#define T_    512
#define I_    512
#define H_    1024
#define TGT_  128
#define TC_   32              // timesteps per chunk
#define CB_   (TC_ * B_)      // 4096 rows per chunk GEMM

// GEMM tiling: 256 threads = 8 waves (wave32). Block tile 128x128, K-chunk 32.
#define BM    128
#define BN    128
#define KC    32
#define KPAD  40              // LDS row padding (bf16 elems) to stagger banks

typedef __attribute__((ext_vector_type(16))) __bf16 v16bf;
typedef __attribute__((ext_vector_type(8)))  __bf16 v8bf;
typedef __attribute__((ext_vector_type(8)))  float  v8f;

__device__ __forceinline__ __bf16 to_bf16(float f) {
  // round-to-nearest-even fp32 -> bf16
  unsigned u = __float_as_uint(f);
  unsigned r = (u + 0x7FFFu + ((u >> 16) & 1u)) >> 16;
  unsigned short h = (unsigned short)r;
  return __builtin_bit_cast(__bf16, h);
}

__device__ __forceinline__ float sigmoidf_(float x) {
  return 1.0f / (1.0f + __expf(-x));
}

// Async global->LDS copy of 16 bytes (CDNA5, tracked by ASYNCcnt).
__device__ __forceinline__ void async_b128_to_lds(unsigned lds_off, const void* g) {
  asm volatile("global_load_async_to_lds_b128 %0, %1, off"
               :: "v"(lds_off), "v"((unsigned long long)(uintptr_t)g)
               : "memory");
}

// Fragment load matching the ISA 16-bit 16x32 A-operand lane layout:
// lanes 0-15 (row M=lane): elems 0-7 = K 0-7,  elems 8-15 = K 16-23
// lanes 16-31 (row M=lane-16): elems 0-7 = K 8-15, elems 8-15 = K 24-31
// B operand (32x16) is staged K-contiguous per column so the same pattern applies.
__device__ __forceinline__ v16bf load_frag(const __bf16* row, int lane) {
  const int kb = ((lane >> 4) & 1) * 8;
  v8bf lo = *(const v8bf*)(row + kb);
  v8bf hi = *(const v8bf*)(row + kb + 16);
  return __builtin_shufflevector(lo, hi,
                                 0, 1, 2, 3, 4, 5, 6, 7,
                                 8, 9, 10, 11, 12, 13, 14, 15);
}

struct GruArgs {
  __bf16* Xc[2];     // [CB_][I_]        bf16 chunk of inputs (time-major)
  __bf16* Wpack[2];  // [I_][3H]         bf16 [Wr|Wu|Wh]
  float*  bpack[2];  // [3H]             f32  [br|bu|bh]
  float*  pre[2];    // [CB_][3H]        f32  precomputed x-projections
  __bf16* Uru[2];    // [H][2H]          bf16 [Ur|Uu]
  __bf16* Uh[2];     // [H][H]           bf16
  __bf16* hbf[2];    // [B_][H]          bf16 state (A operand for phase 1)
  float*  hf[2];     // [B_][H]          f32 state
  __bf16* rh[2];     // [B_][H]          bf16 r*h (A operand for phase 2)
  float*  ubuf[2];   // [B_][H]          f32 update gate
  __bf16* hs;        // [CB_][2H]        bf16 concat(h_f, h_b) for output GEMM
  __bf16* Wo;        // [2H][TGT]        bf16
  const float* bo;   // [TGT]
  float*  out;       // [T][B][TGT]      f32 (d_out)
};

// MODE 0: pre = Xc @ Wpack + bpack           (M=CB_, N=3H, K=I)
// MODE 1: ru  = hbf @ [Ur|Uu]; r,u = sigmoid(pre + ru); rh = r*h; store u
// MODE 2: c = tanh(pre_h + rh @ Uh); h = u*h+(1-u)*c; update hf/hbf/hs
// MODE 3: out = sigmoid(hs @ Wo + bo)        (M=CB_, N=TGT, K=2H)
template <int MODE>
__global__ __launch_bounds__(256) void gru_wmma_gemm(GruArgs P, int step) {
  constexpr int M = (MODE == 0 || MODE == 3) ? CB_ : B_;
  constexpr int N = (MODE == 0) ? 3 * H_ : (MODE == 1) ? 2 * H_
                    : (MODE == 2) ? H_ : TGT_;
  constexpr int K = (MODE == 0) ? I_ : (MODE == 3) ? 2 * H_ : H_;
  (void)M;

  const int dir = blockIdx.z;
  const __bf16* Ag;
  const __bf16* Bg;
  if constexpr (MODE == 0)      { Ag = P.Xc[dir];  Bg = P.Wpack[dir]; }
  else if constexpr (MODE == 1) { Ag = P.hbf[dir]; Bg = P.Uru[dir]; }
  else if constexpr (MODE == 2) { Ag = P.rh[dir];  Bg = P.Uh[dir]; }
  else                          { Ag = P.hs;       Bg = P.Wo; }

  const int tid  = threadIdx.x;
  const int lane = tid & 31;
  const int wave = tid >> 5;
  const int bn0  = blockIdx.x * BN;
  const int bm0  = blockIdx.y * BM;

  // Ping-pong double buffer: async-stage chunk k+1 while computing chunk k.
  __shared__ __bf16 As[2][BM][KPAD];   // [buf][row m][k]
  __shared__ __bf16 Bs[2][BN][KPAD];   // [buf][col n][k] (transposed)

  // Stage one 128x32 A tile (async b128, ASYNCcnt) and one 32x128 B tile
  // (coalesced 16B global reads, transposed scalar LDS writes).
  auto stage = [&](int buf, int k0) {
#pragma unroll
    for (int it = 0; it < 2; ++it) {
      const int c   = tid + it * 256;     // 0..511
      const int row = c >> 2;             // 0..127
      const int seg = c & 3;              // 16B segment of the 64B row
      const __bf16* g = Ag + (size_t)(bm0 + row) * K + (k0 + seg * 8);
      async_b128_to_lds((unsigned)(uintptr_t)(&As[buf][row][seg * 8]), g);
    }
#pragma unroll
    for (int it = 0; it < 2; ++it) {
      const int c    = tid + it * 256;    // 0..511
      const int k    = c >> 4;            // 0..31
      const int nseg = c & 15;            // 8 columns per chunk
      v8bf vals = *(const v8bf*)(Bg + (size_t)(k0 + k) * N + bn0 + nseg * 8);
#pragma unroll
      for (int j = 0; j < 8; ++j) Bs[buf][nseg * 8 + j][k] = vals[j];
    }
  };

  v8f zero;
#pragma unroll
  for (int e = 0; e < 8; ++e) zero[e] = 0.0f;
  v8f acc[8];
#pragma unroll
  for (int j = 0; j < 8; ++j) acc[j] = zero;

  stage(0, 0);
  asm volatile("s_wait_asynccnt 0" ::: "memory");
  __syncthreads();

  for (int k0 = 0; k0 < K; k0 += KC) {
    const int buf = (k0 / KC) & 1;
    if (k0 + KC < K) {
      if (k0 + 2 * KC < K) {  // hint the chunk after next (global_prefetch_b8)
        __builtin_prefetch(Ag + (size_t)(bm0 + (tid >> 1)) * K + k0 + 2 * KC,
                           0, 1);
        __builtin_prefetch(Bg + (size_t)(k0 + 2 * KC + (tid >> 3)) * N + bn0,
                           0, 1);
      }
      stage(buf ^ 1, k0 + KC);   // async loads overlap the WMMAs below
    }

    // Preload all fragments, then issue 8 back-to-back WMMAs (one dscnt wait).
    const v16bf afrag = load_frag(&As[buf][wave * 16 + (lane & 15)][0], lane);
    v16bf bfrag[8];
#pragma unroll
    for (int nt = 0; nt < 8; ++nt)
      bfrag[nt] = load_frag(&Bs[buf][nt * 16 + (lane & 15)][0], lane);
#pragma unroll
    for (int nt = 0; nt < 8; ++nt) {
      acc[nt] = __builtin_amdgcn_wmma_f32_16x16x32_bf16(
          false, afrag, false, bfrag[nt], (short)0, acc[nt], false, false);
    }

    asm volatile("s_wait_asynccnt 0" ::: "memory");
    __syncthreads();
  }

  // ---- Epilogue. C/D layout: lanes 0-15: VGPR i -> M=i; lanes 16-31: M=8+i.
#pragma unroll
  for (int nt = 0; nt < 8; ++nt) {
    const int n  = bn0 + nt * 16 + (lane & 15);
    if (n >= N) continue;
    const int mb = bm0 + wave * 16 + ((lane >> 4) * 8);
#pragma unroll
    for (int i = 0; i < 8; ++i) {
      const int m = mb + i;
      const float v = acc[nt][i];
      if constexpr (MODE == 0) {
        P.pre[dir][(size_t)m * (3 * H_) + n] = v + P.bpack[dir][n];
      } else if constexpr (MODE == 1) {
        const size_t prow = ((size_t)step * B_ + m) * (3 * H_);
        if (n < H_) {
          const float r = sigmoidf_(P.pre[dir][prow + n] + v);
          P.rh[dir][(size_t)m * H_ + n] =
              to_bf16(r * P.hf[dir][(size_t)m * H_ + n]);
        } else {
          const int nu = n - H_;
          const float u = sigmoidf_(P.pre[dir][prow + H_ + nu] + v);
          P.ubuf[dir][(size_t)m * H_ + nu] = u;
        }
      } else if constexpr (MODE == 2) {
        const size_t prow = ((size_t)step * B_ + m) * (3 * H_);
        const float c  = tanhf(P.pre[dir][prow + 2 * H_ + n] + v);
        const float u  = P.ubuf[dir][(size_t)m * H_ + n];
        const float h  = P.hf[dir][(size_t)m * H_ + n];
        const float hn = u * h + (1.0f - u) * c;
        P.hf[dir][(size_t)m * H_ + n]  = hn;
        P.hbf[dir][(size_t)m * H_ + n] = to_bf16(hn);
        P.hs[((size_t)step * B_ + m) * (2 * H_) + (size_t)dir * H_ + n] =
            to_bf16(hn);
      } else {  // MODE 3: step == t0 (chunk base timestep)
        const float o = sigmoidf_(v + P.bo[n]);
        P.out[((size_t)step * B_ + m) * TGT_ + n] = o;
      }
    }
  }
}

// fp32 -> bf16 strided pack (weight concat along N)
__global__ void cvt_strided_bf16(__bf16* dst, int dld, const float* src,
                                 int sld, int rows, int cols) {
  const int idx = blockIdx.x * blockDim.x + threadIdx.x;
  if (idx >= rows * cols) return;
  const int r = idx / cols, c = idx % cols;
  dst[(size_t)r * dld + c] = to_bf16(src[(size_t)r * sld + c]);
}

// Gather chunk of TC_ timesteps from [B,T,I] into time-major bf16 [TC_*B, I]
__global__ void cvt_x_chunk(__bf16* dstF, __bf16* dstBk, const float* srcF,
                            const float* srcBk, int t0) {
  const float* s = blockIdx.y ? srcBk : srcF;
  __bf16*      d = blockIdx.y ? dstBk : dstF;
  const int idx = blockIdx.x * 256 + threadIdx.x;   // < CB_*I_
  const int i   = idx & (I_ - 1);
  const int row = idx >> 9;                         // dt*B + b
  const int dt  = row >> 7;
  const int b   = row & (B_ - 1);
  d[idx] = to_bf16(s[((size_t)b * T_ + (t0 + dt)) * I_ + i]);
}

__global__ void init_h(float* h0, float* h1, __bf16* hb0, __bf16* hb1) {
  const int idx = blockIdx.x * blockDim.x + threadIdx.x;
  if (idx >= B_ * H_) return;
  h0[idx] = 0.0f; h1[idx] = 0.0f;
  hb0[idx] = to_bf16(0.0f); hb1[idx] = to_bf16(0.0f);
}

extern "C" void kernel_launch(void* const* d_in, const int* in_sizes, int n_in,
                              void* d_out, int out_size, void* d_ws,
                              size_t ws_size, hipStream_t stream) {
  (void)in_sizes; (void)n_in; (void)out_size; (void)ws_size;

  const float* xf = (const float*)d_in[0];
  const float* xb = (const float*)d_in[1];
  const float* Wr[2]  = {(const float*)d_in[2],  (const float*)d_in[11]};
  const float* Ur[2]  = {(const float*)d_in[3],  (const float*)d_in[12]};
  const float* br[2]  = {(const float*)d_in[4],  (const float*)d_in[13]};
  const float* Wu[2]  = {(const float*)d_in[5],  (const float*)d_in[14]};
  const float* Uu[2]  = {(const float*)d_in[6],  (const float*)d_in[15]};
  const float* bu[2]  = {(const float*)d_in[7],  (const float*)d_in[16]};
  const float* Wh[2]  = {(const float*)d_in[8],  (const float*)d_in[17]};
  const float* UhS[2] = {(const float*)d_in[9],  (const float*)d_in[18]};
  const float* bh[2]  = {(const float*)d_in[10], (const float*)d_in[19]};
  const float* WoS = (const float*)d_in[20];
  const float* boS = (const float*)d_in[21];

  char* wp = (char*)d_ws;
  auto take = [&](size_t bytes) {
    void* p = (void*)wp;
    wp += (bytes + 255) & ~(size_t)255;
    return p;
  };

  GruArgs P;
  for (int d = 0; d < 2; ++d) {
    P.Wpack[d] = (__bf16*)take((size_t)I_ * 3 * H_ * 2);
    P.Uru[d]   = (__bf16*)take((size_t)H_ * 2 * H_ * 2);
    P.Uh[d]    = (__bf16*)take((size_t)H_ * H_ * 2);
    P.bpack[d] = (float*) take((size_t)3 * H_ * 4);
    P.Xc[d]    = (__bf16*)take((size_t)CB_ * I_ * 2);
    P.pre[d]   = (float*) take((size_t)CB_ * 3 * H_ * 4);
    P.hf[d]    = (float*) take((size_t)B_ * H_ * 4);
    P.hbf[d]   = (__bf16*)take((size_t)B_ * H_ * 2);
    P.rh[d]    = (__bf16*)take((size_t)B_ * H_ * 2);
    P.ubuf[d]  = (float*) take((size_t)B_ * H_ * 4);
  }
  P.hs  = (__bf16*)take((size_t)CB_ * 2 * H_ * 2);
  P.Wo  = (__bf16*)take((size_t)2 * H_ * TGT_ * 2);
  P.bo  = boS;
  P.out = (float*)d_out;

  auto cvt = [&](__bf16* dst, int dld, const float* src, int sld, int rows,
                 int cols) {
    const int total = rows * cols;
    cvt_strided_bf16<<<(total + 255) / 256, 256, 0, stream>>>(dst, dld, src,
                                                              sld, rows, cols);
  };

  // Pack weights to bf16 (concat along N so r/u and r|u|h share one GEMM).
  for (int d = 0; d < 2; ++d) {
    cvt(P.Wpack[d] + 0,      3 * H_, Wr[d], H_, I_, H_);
    cvt(P.Wpack[d] + H_,     3 * H_, Wu[d], H_, I_, H_);
    cvt(P.Wpack[d] + 2 * H_, 3 * H_, Wh[d], H_, I_, H_);
    cvt(P.Uru[d] + 0,  2 * H_, Ur[d], H_, H_, H_);
    cvt(P.Uru[d] + H_, 2 * H_, Uu[d], H_, H_, H_);
    cvt(P.Uh[d], H_, UhS[d], H_, H_, H_);
    hipMemcpyAsync(P.bpack[d],          br[d], H_ * 4,
                   hipMemcpyDeviceToDevice, stream);
    hipMemcpyAsync(P.bpack[d] + H_,     bu[d], H_ * 4,
                   hipMemcpyDeviceToDevice, stream);
    hipMemcpyAsync(P.bpack[d] + 2 * H_, bh[d], H_ * 4,
                   hipMemcpyDeviceToDevice, stream);
  }
  cvt(P.Wo, TGT_, WoS, TGT_, 2 * H_, TGT_);

  init_h<<<(B_ * H_ + 255) / 256, 256, 0, stream>>>(P.hf[0], P.hf[1],
                                                    P.hbf[0], P.hbf[1]);

  for (int ch = 0; ch < T_ / TC_; ++ch) {
    const int t0 = ch * TC_;
    cvt_x_chunk<<<dim3((CB_ * I_) / 256, 2), 256, 0, stream>>>(
        P.Xc[0], P.Xc[1], xf, xb, t0);
    // Input projections for this chunk, both directions (z=2).
    gru_wmma_gemm<0><<<dim3(3 * H_ / BN, CB_ / BM, 2), 256, 0, stream>>>(P, 0);
    // Sequential recurrence: two dependent WMMA GEMMs per step.
    for (int dt = 0; dt < TC_; ++dt) {
      gru_wmma_gemm<1><<<dim3(2 * H_ / BN, 1, 2), 256, 0, stream>>>(P, dt);
      gru_wmma_gemm<2><<<dim3(H_ / BN, 1, 2), 256, 0, stream>>>(P, dt);
    }
    // Fused output projection + sigmoid for the chunk.
    gru_wmma_gemm<3><<<dim3(TGT_ / BN, CB_ / BM, 1), 256, 0, stream>>>(P, t0);
  }
}
